// TailGNN_10866267259409
// MI455X (gfx1250) — compile-verified, matrix operand order checked by reading, not run on hardware
//
#include <hip/hip_runtime.h>
#include <hip/hip_bf16.h>

// ---------------------------------------------------------------------------
// TailGNN forward on MI455X (gfx1250): bf16 WMMA GEMMs + fused elementwise.
// LDS tiles staged in packed-fragment layout -> ds_load_b128 fragment reads.
// fp32->bf16x2 packing: RNE bias + single v_perm_b32 splice (32-bit ops only).
// ---------------------------------------------------------------------------

#define NN    8192
#define NF    256
#define NH    128
#define NC    16
#define LRELU 0.2f
#define EPSN  1e-12f

typedef __attribute__((ext_vector_type(16))) __bf16 v16bf;
typedef __attribute__((ext_vector_type(8)))  float  v8f;

union BFrag { v16bf v; unsigned u[8]; };

// pack two floats -> one dword of two bf16 (lo = even K, hi = odd K), RNE.
// 32-bit arithmetic only: avoids v_mov_b16 sub-register churn; final splice is
// one v_perm_b32 selecting the high halves (sel 0x07060302: lo.b2,lo.b3,hi.b2,hi.b3).
__device__ __forceinline__ unsigned pk2(float lo, float hi) {
    unsigned ul = __float_as_uint(lo);
    unsigned uh = __float_as_uint(hi);
    ul += 0x7FFFu + ((ul >> 16) & 1u);   // round-to-nearest-even
    uh += 0x7FFFu + ((uh >> 16) & 1u);
    return __builtin_amdgcn_perm(uh, ul, 0x07060302u);
}

// ---------------------------------------------------------------------------
// Row sums of adj (adj >= 0 so |.| == .) : deg[row] = sum_j adj[row, j]
// ---------------------------------------------------------------------------
__global__ void rowsum_kernel(const float* __restrict__ adj,
                              float* __restrict__ deg, int n) {
    __shared__ float red[256];
    const int row = blockIdx.x;
    const float* a = adj + (size_t)row * n;
    float s = 0.f;
    for (int c = threadIdx.x; c < n; c += 256) s += a[c];
    red[threadIdx.x] = s;
    __syncthreads();
    for (int off = 128; off > 0; off >>= 1) {
        if (threadIdx.x < off) red[threadIdx.x] += red[threadIdx.x + off];
        __syncthreads();
    }
    if (threadIdx.x == 0) deg[row] = red[0];
}

// ---------------------------------------------------------------------------
// C[M,N] = A[M,K] @ B[K,N], row-major fp32 in / fp32 out, bf16 WMMA inside.
// Block = 256 threads = 8 waves. Tile: 128(M) x BN(N), K-step 32.
// LDS holds packed bf16-pair dwords indexed by [row|col][kpair] so each lane's
// fragment is two contiguous uint4 (ds_load_b128) reads.
// Software pipeline: next global tile fetched to registers during WMMA.
// Requires: M % 128 == 0, N % BN == 0, K % 32 == 0 (true for all shapes here).
// ---------------------------------------------------------------------------
template<int BN>
__global__ void gemm_bf16_wmma(const float* __restrict__ A,
                               const float* __restrict__ B,
                               float* __restrict__ C,
                               int M, int N, int K) {
    constexpr int BM  = 128, BK = 32;
    constexpr int NT  = BN / 16;
    constexpr int LDP = 20;                  // padded row stride in dwords (80B)
    constexpr int NGB = (BK / 2) * (BN / 4); // B staging groups (<= 256)
    __shared__ unsigned Apack[BM][LDP];      // [row][kpair]  kpair = K/2
    __shared__ unsigned Bpack[BN][LDP];      // [col][kpair]

    const int tid  = threadIdx.x;
    const int wave = tid >> 5;
    const int lane = tid & 31;
    const int half = lane >> 4;              // 0: kp 0-3/8-11 | 1: kp 4-7/12-15
    const int l15  = lane & 15;
    const int tileM = blockIdx.y * BM;
    const int tileN = blockIdx.x * BN;

    // B staging mapping: thread -> (kpair row, 4-column group)
    const int kp = tid / (BN / 4);
    const int cb = (tid % (BN / 4)) * 4;
    const bool bact = (NGB >= 256) || (tid < NGB);   // constexpr-folds for BN=64

    v8f acc[NT];
#pragma unroll
    for (int t = 0; t < NT; ++t) acc[t] = (v8f){0,0,0,0,0,0,0,0};

    float4 a4[4], b0, b1;

    auto fetch = [&](int k0) {
#pragma unroll
        for (int i = 0; i < 4; ++i) {
            int q = tid + i * 256;
            int r = q >> 3, c = (q & 7) * 4;   // 8 float4 per 32-wide row
            a4[i] = *reinterpret_cast<const float4*>(
                &A[(size_t)(tileM + r) * K + k0 + c]);
        }
        if (bact) {
            b0 = *reinterpret_cast<const float4*>(
                &B[(size_t)(k0 + 2 * kp) * N + tileN + cb]);
            b1 = *reinterpret_cast<const float4*>(
                &B[(size_t)(k0 + 2 * kp + 1) * N + tileN + cb]);
        }
    };
    auto stage = [&]() {
#pragma unroll
        for (int i = 0; i < 4; ++i) {
            int q = tid + i * 256;
            int r = q >> 3, c2 = (q & 7) * 2;  // 2 packed dwords per float4
            Apack[r][c2]     = pk2(a4[i].x, a4[i].y);
            Apack[r][c2 + 1] = pk2(a4[i].z, a4[i].w);
        }
        if (bact) {
            Bpack[cb + 0][kp] = pk2(b0.x, b1.x);
            Bpack[cb + 1][kp] = pk2(b0.y, b1.y);
            Bpack[cb + 2][kp] = pk2(b0.z, b1.z);
            Bpack[cb + 3][kp] = pk2(b0.w, b1.w);
        }
    };

    fetch(0);
    for (int k0 = 0; k0 < K; k0 += BK) {
        stage();
        __syncthreads();
        if (k0 + BK < K) fetch(k0 + BK);       // overlap global with WMMA

        // ---- A fragment: two b128 LDS reads ----
        const unsigned* arow = &Apack[wave * 16 + l15][0];
        uint4 alo = *reinterpret_cast<const uint4*>(arow + half * 4);
        uint4 ahi = *reinterpret_cast<const uint4*>(arow + 8 + half * 4);
        BFrag af;
        af.u[0] = alo.x; af.u[1] = alo.y; af.u[2] = alo.z; af.u[3] = alo.w;
        af.u[4] = ahi.x; af.u[5] = ahi.y; af.u[6] = ahi.z; af.u[7] = ahi.w;

#pragma unroll
        for (int t = 0; t < NT; ++t) {
            const unsigned* brow = &Bpack[t * 16 + l15][0];
            uint4 blo = *reinterpret_cast<const uint4*>(brow + half * 4);
            uint4 bhi = *reinterpret_cast<const uint4*>(brow + 8 + half * 4);
            BFrag bf;
            bf.u[0] = blo.x; bf.u[1] = blo.y; bf.u[2] = blo.z; bf.u[3] = blo.w;
            bf.u[4] = bhi.x; bf.u[5] = bhi.y; bf.u[6] = bhi.z; bf.u[7] = bhi.w;
            acc[t] = __builtin_amdgcn_wmma_f32_16x16x32_bf16(
                false, af.v, false, bf.v, (short)0, acc[t], false, false);
        }
        __syncthreads();
    }

    // ---- store: VGPR r holds M = wave*16 + half*8 + r, N = tileN+t*16+l15 ----
#pragma unroll
    for (int t = 0; t < NT; ++t) {
        int col = tileN + t * 16 + l15;
#pragma unroll
        for (int r = 0; r < 8; ++r) {
            int row = tileM + wave * 16 + half * 8 + r;
            C[(size_t)row * N + col] = acc[t][r];
        }
    }
}

// ---------------------------------------------------------------------------
// Elementwise kernels
// ---------------------------------------------------------------------------
__global__ void scale_rows_kernel(const float* __restrict__ P,
                                  const float* __restrict__ deg,
                                  float* __restrict__ nb, int d, int total) {
    int i = blockIdx.x * 256 + threadIdx.x;
    if (i >= total) return;
    int row = i / d;
    nb[i] = P[i] / fmaxf(deg[row], EPSN);
}

// out = x + (leaky(ga+gb)+1) * r[col] + leaky(ba+bb) - nb
__global__ void relation_kernel(const float* __restrict__ x,
                                const float* __restrict__ nb,
                                const float* __restrict__ ga,
                                const float* __restrict__ gb,
                                const float* __restrict__ ba,
                                const float* __restrict__ bb,
                                const float* __restrict__ r,
                                float* __restrict__ out, int d, int total) {
    int i = blockIdx.x * 256 + threadIdx.x;
    if (i >= total) return;
    int col = i % d;
    float g = ga[i] + gb[i]; g = (g > 0.f) ? g : LRELU * g; g += 1.f;
    float b = ba[i] + bb[i]; b = (b > 0.f) ? b : LRELU * b;
    out[i] = x[i] + g * r[col] + b - nb[i];
}

// out = ((adj@s) + s + (rel@w)) / (deg + 2), optional ELU
__global__ void combine_kernel(const float* __restrict__ tAS,
                               const float* __restrict__ s,
                               const float* __restrict__ tOW,
                               const float* __restrict__ deg,
                               float* __restrict__ out,
                               int d, int total, int do_elu) {
    int i = blockIdx.x * 256 + threadIdx.x;
    if (i >= total) return;
    int row = i / d;
    float h = (tAS[i] + s[i] + tOW[i]) / (deg[row] + 2.f);
    if (do_elu) h = (h > 0.f) ? h : expm1f(h);
    out[i] = h;
}

__global__ void logsoftmax_kernel(const float* __restrict__ x,
                                  float* __restrict__ lp, int d, int rows) {
    int row = blockIdx.x * 256 + threadIdx.x;
    if (row >= rows) return;
    const float* xr = x + (size_t)row * d;
    float m = -3.4e38f;
    for (int c = 0; c < d; ++c) m = fmaxf(m, xr[c]);
    float se = 0.f;
    for (int c = 0; c < d; ++c) se += expf(xr[c] - m);
    float lse = m + logf(se);
    float* o = lp + (size_t)row * d;
    for (int c = 0; c < d; ++c) o[c] = xr[c] - lse;
}

// ---------------------------------------------------------------------------
// Host-side orchestration
// ---------------------------------------------------------------------------
extern "C" void kernel_launch(void* const* d_in, const int* in_sizes, int n_in,
                              void* d_out, int out_size, void* d_ws, size_t ws_size,
                              hipStream_t stream) {
    const float* x    = (const float*)d_in[0];
    const float* adj  = (const float*)d_in[1];
    const float* r1g1 = (const float*)d_in[2];
    const float* r1g2 = (const float*)d_in[3];
    const float* r1b1 = (const float*)d_in[4];
    const float* r1b2 = (const float*)d_in[5];
    const float* r1r  = (const float*)d_in[6];
    const float* w1   = (const float*)d_in[7];
    const float* r2g1 = (const float*)d_in[8];
    const float* r2g2 = (const float*)d_in[9];
    const float* r2b1 = (const float*)d_in[10];
    const float* r2b2 = (const float*)d_in[11];
    const float* r2r  = (const float*)d_in[12];
    const float* w2   = (const float*)d_in[13];

    // workspace carve-up (floats); layer 2 reuses the layer-1 buffers
    float* ws = (float*)d_ws;
    size_t o = 0;
    float* deg = ws + o; o += NN;
    float* P   = ws + o; o += (size_t)NN * NF;
    float* nb  = ws + o; o += (size_t)NN * NF;
    float* ga  = ws + o; o += (size_t)NN * NF;
    float* gb  = ws + o; o += (size_t)NN * NF;
    float* ba  = ws + o; o += (size_t)NN * NF;
    float* bb  = ws + o; o += (size_t)NN * NF;
    float* s1  = ws + o; o += (size_t)NN * NH;
    float* tAS = ws + o; o += (size_t)NN * NH;
    float* tOW = ws + o; o += (size_t)NN * NH;
    float* x1  = ws + o; o += (size_t)NN * NH;

    // outputs: (x2, log_softmax, out1, out2) flat in return order
    float* outp = (float*)d_out;
    float* x2o  = outp;
    float* lpo  = outp + (size_t)NN * NC;
    float* out1 = outp + (size_t)NN * NC * 2;
    float* out2 = out1 + (size_t)NN * NF;

    const dim3 blk(256);
    auto G64 = [&](const float* A, const float* B, float* C, int M, int Nc, int K) {
        gemm_bf16_wmma<64><<<dim3(Nc / 64, M / 128), blk, 0, stream>>>(A, B, C, M, Nc, K);
    };
    auto G16 = [&](const float* A, const float* B, float* C, int M, int Nc, int K) {
        gemm_bf16_wmma<16><<<dim3(Nc / 16, M / 128), blk, 0, stream>>>(A, B, C, M, Nc, K);
    };
    auto EW = [&](int total) { return dim3((total + 255) / 256); };

    // ---------------- layer 1 (d = 256 -> 128) ----------------
    rowsum_kernel<<<NN, blk, 0, stream>>>(adj, deg, NN);
    G64(adj, x, P, NN, NF, NN);                                        // adj @ x
    scale_rows_kernel<<<EW(NN*NF), blk, 0, stream>>>(P, deg, nb, NF, NN*NF);
    G64(x,  r1g1, ga, NN, NF, NF);
    G64(nb, r1g2, gb, NN, NF, NF);
    G64(x,  r1b1, ba, NN, NF, NF);
    G64(nb, r1b2, bb, NN, NF, NF);
    relation_kernel<<<EW(NN*NF), blk, 0, stream>>>(x, nb, ga, gb, ba, bb,
                                                   r1r, out1, NF, NN*NF);
    G64(x, w1, s1, NN, NH, NF);                                        // x @ w1
    G64(adj, s1, tAS, NN, NH, NN);                                     // adj @ s
    G64(out1, w1, tOW, NN, NH, NF);                                    // out1 @ w1
    combine_kernel<<<EW(NN*NH), blk, 0, stream>>>(tAS, s1, tOW, deg, x1,
                                                  NH, NN*NH, /*elu=*/1);

    // ---------------- layer 2 (d = 128 -> 16) ----------------
    G64(adj, x1, P, NN, NH, NN);                                       // adj @ x1
    scale_rows_kernel<<<EW(NN*NH), blk, 0, stream>>>(P, deg, nb, NH, NN*NH);
    G64(x1, r2g1, ga, NN, NH, NH);
    G64(nb, r2g2, gb, NN, NH, NH);
    G64(x1, r2b1, ba, NN, NH, NH);
    G64(nb, r2b2, bb, NN, NH, NH);
    relation_kernel<<<EW(NN*NH), blk, 0, stream>>>(x1, nb, ga, gb, ba, bb,
                                                   r2r, out2, NH, NN*NH);
    G16(x1, w2, s1, NN, NC, NH);                                       // x1 @ w2
    G16(adj, s1, tAS, NN, NC, NN);                                     // adj @ s2
    G16(out2, w2, tOW, NN, NC, NH);                                    // out2 @ w2
    combine_kernel<<<EW(NN*NC), blk, 0, stream>>>(tAS, s1, tOW, deg, x2o,
                                                  NC, NN*NC, /*elu=*/0);
    logsoftmax_kernel<<<EW(NN), blk, 0, stream>>>(x2o, lpo, NC, NN);
}
